// ChebNetGCN_43361989820616
// MI455X (gfx1250) — matrix operand order repaired and checked by reading, not compiled, not run on hardware
//
#include <hip/hip_runtime.h>

typedef __bf16 bf16_t;
typedef __attribute__((ext_vector_type(8)))  __bf16 v8bf;
typedef __attribute__((ext_vector_type(16))) __bf16 v16bf;
typedef __attribute__((ext_vector_type(8)))  float  v8f;
typedef __attribute__((ext_vector_type(4)))  float  v4f;
typedef __attribute__((ext_vector_type(4)))  unsigned int v4u;
typedef __attribute__((ext_vector_type(8)))  int v8i;
typedef __attribute__((ext_vector_type(4)))  int v4i;

constexpr int BB    = 512;   // batch
constexpr int NODES = 128;   // graph nodes
constexpr int F_IN  = 512;
constexpr int HID   = 512;
constexpr int C_OUTc= 256;
constexpr int NCc   = 8;

#if __has_builtin(__builtin_amdgcn_tensor_load_to_lds) && __has_builtin(__builtin_amdgcn_s_wait_tensorcnt)
#define HAVE_TDM 1
#else
#define HAVE_TDM 0
#endif

// build a v16bf WMMA fragment from two 16-byte chunks
__device__ __forceinline__ v16bf frag2(const void* p0, const void* p1) {
  v8bf lo = *(const v8bf*)p0;
  v8bf hi = *(const v8bf*)p1;
  return __builtin_shufflevector(lo, hi, 0,1,2,3,4,5,6,7,8,9,10,11,12,13,14,15);
}
__device__ __forceinline__ v8f wmma_bf16(v16bf a, v16bf b, v8f c) {
  return __builtin_amdgcn_wmma_f32_16x16x32_bf16(false, a, false, b, (short)0, c, false, false);
}

// ---------------------------------------------------------------- utilities
__global__ void f32_to_bf16_kernel(const float* __restrict__ src,
                                   bf16_t* __restrict__ dst, int n) {
  int i = blockIdx.x * 256 + threadIdx.x;
  if (i < n) dst[i] = (bf16_t)src[i];
}

// x [B,N,F] f32 -> x_nm bf16 (same layout) + x_fm bf16 [B,F,N]
__global__ void x_prep_kernel(const float* __restrict__ x,
                              bf16_t* __restrict__ xnm, bf16_t* __restrict__ xfm) {
  long long idx = (long long)blockIdx.x * 256 + threadIdx.x;
  long long total = (long long)BB * NODES * F_IN;
  if (idx >= total) return;
  int f = (int)(idx % F_IN);
  long long r = idx / F_IN;
  int n = (int)(r % NODES);
  int b = (int)(r / NODES);
  bf16_t v = (bf16_t)x[idx];
  xnm[idx] = v;
  xfm[((long long)b * F_IN + f) * NODES + n] = v;
}

// Lhat = -D^{-1/2} Asym D^{-1/2}  (symmetric, zero diagonal)
__global__ void normalize_kernel(const float* __restrict__ A,
                                 bf16_t* __restrict__ L) {
  int b = blockIdx.x;
  const float* Ab = A + (size_t)b * NODES * NODES;
  __shared__ float dinv[NODES];
  int i = threadIdx.x;                      // 128 threads
  float s = 0.f;
  for (int j = 0; j < NODES; ++j) {
    if (j != i) s += fmaxf(Ab[i * NODES + j], 0.f) + fmaxf(Ab[j * NODES + i], 0.f);
  }
  dinv[i] = rsqrtf(s + 1e-10f);
  __syncthreads();
  bf16_t* Lb = L + (size_t)b * NODES * NODES;
  for (int row = 0; row < NODES; ++row) {
    int j = i;
    float v = 0.f;
    if (j != row) {
      float a = fmaxf(Ab[row * NODES + j], 0.f) + fmaxf(Ab[j * NODES + row], 0.f);
      v = -a * dinv[row] * dinv[j];
    }
    Lb[row * NODES + j] = (bf16_t)v;
  }
}

// --------------------------------------------------- TDM: L tile -> LDS
// 32 rows x 128 cols bf16, 4-DWORD pad per 64-DWORD row -> 272 B rows
constexpr int ROWL = 272;

__device__ __forceinline__ void tdm_load_L(const bf16_t* gsrc, unsigned lds_off) {
#if HAVE_TDM
  unsigned long long ga = (unsigned long long)gsrc;
  v4u g0 = { 1u, lds_off,
             (unsigned)(ga & 0xFFFFFFFFu),
             (unsigned)(((ga >> 32) & 0x1FFFFFFu) | (2u << 30)) };
  v8i g1 = { (int)0x07510000,   // data_size=2B | pad_en | pad_interval=5(64DW) | pad_amount=3(4DW)
             (int)(128u << 16), // tensor_dim0 = 128
             (int)(128u << 16), // tensor_dim1 = 128
             (int)(128u << 16), // tile_dim0 = 128
             (int)32,           // tile_dim1 = 32
             (int)128,          // tensor_dim0_stride = 128
             0, 0 };
  v4i z4 = {0, 0, 0, 0};
#if __clang_major__ >= 23
  v8i z8 = {0, 0, 0, 0, 0, 0, 0, 0};
  __builtin_amdgcn_tensor_load_to_lds(g0, g1, z4, z4, z8, 0);
#else
  __builtin_amdgcn_tensor_load_to_lds(g0, g1, z4, z4, 0);
#endif
#endif
}

__device__ __forceinline__ v16bf bfrag_L(const char* p, int nloc, int k0, int half) {
  const char* base = p + nloc * ROWL + k0 * 2 + 16 * half;
  return frag2(base, base + 32);
}

// ------------------------------------------------------------- cheb matmul
// out[f,m] = alpha * sum_k Xt[f,k]*L[m,k]  (+ beta * Prev_nm[m,f])
// Block: 4 waves, same (b, n0); L tile staged once via TDM; 2x2 tiles/wave.
template <bool HAS_PREV, bool WRITE_FM>
__global__ __launch_bounds__(128)
void cheb_wmma_kernel(const bf16_t* __restrict__ Xt,    // [B,F,N] feature-major
                      const bf16_t* __restrict__ L,     // [B,N,N] symmetric
                      const bf16_t* __restrict__ Prev,  // [B,N,F] node-major
                      bf16_t* __restrict__ OutFM,       // [B,F,N]
                      bf16_t* __restrict__ OutNM,       // [B,N,F]
                      int Fdim, float alpha, float beta) {
  int wave = threadIdx.x >> 5;
  int lane = threadIdx.x & 31;
  int half = lane >> 4;
  int lm   = lane & 15;

  int fB  = Fdim / 128;                     // 4 (f-tiles of 32, 4 per block)
  int bid = blockIdx.x;
  int b   = bid / (4 * fB);                 // NODES/32 = 4
  int rem = bid % (4 * fB);
  int n0  = (rem / fB) * 32;
  int f0  = ((rem % fB) * 4 + wave) * 32;

  size_t bstride = (size_t)Fdim * NODES;
  const bf16_t* Xb = Xt + (size_t)b * bstride;
  const bf16_t* Ls = L + (size_t)b * NODES * NODES + (size_t)n0 * NODES;

  __shared__ __align__(16) char panel[32 * ROWL];      // 8.7 KB
  unsigned panel_off = (unsigned)(unsigned long long)(void*)panel;

#if HAVE_TDM
  if (threadIdx.x < 32) {
    tdm_load_L(Ls, panel_off);
    __builtin_amdgcn_s_wait_tensorcnt(0);
  }
#else
  {
    int row = threadIdx.x >> 2, seg = threadIdx.x & 3;
#pragma unroll
    for (int j = 0; j < 4; ++j)
      *(v8bf*)(panel + row * ROWL + seg * 64 + j * 16) =
          *(const v8bf*)(Ls + (size_t)row * NODES + seg * 32 + j * 8);
  }
#endif
  __syncthreads();

  v8f acc00 = {}, acc01 = {}, acc10 = {}, acc11 = {};
#pragma unroll
  for (int k0 = 0; k0 < NODES; k0 += 32) {
    const bf16_t* ar0 = Xb + (size_t)(f0 + lm) * NODES + k0 + 8 * half;
    const bf16_t* ar1 = ar0 + 16 * NODES;
    v16bf a0 = frag2(ar0, ar0 + 16);
    v16bf a1 = frag2(ar1, ar1 + 16);
    v16bf b0 = bfrag_L(panel, lm, k0, half);
    v16bf b1 = bfrag_L(panel, 16 + lm, k0, half);
    acc00 = wmma_bf16(a0, b0, acc00);
    acc01 = wmma_bf16(a0, b1, acc01);
    acc10 = wmma_bf16(a1, b0, acc10);
    acc11 = wmma_bf16(a1, b1, acc11);
  }

  bf16_t* OFb = WRITE_FM ? OutFM + (size_t)b * bstride : nullptr;
  bf16_t* ONb = OutNM + (size_t)b * bstride;
  const bf16_t* Pb = HAS_PREV ? (Prev + (size_t)b * bstride) : nullptr;

#define CHEB_STORE(ACC, FI, NI) {                                          \
    int fb = f0 + (FI)*16 + 8 * half;                                      \
    int n  = n0 + (NI)*16 + lm;                                            \
    v8bf pv{}, ov;                                                         \
    if (HAS_PREV) pv = *(const v8bf*)(Pb + (size_t)n * Fdim + fb);         \
    _Pragma("unroll")                                                      \
    for (int r = 0; r < 8; ++r) {                                          \
      float v = alpha * (ACC)[r];                                          \
      if (HAS_PREV) v += beta * (float)pv[r];                              \
      ov[r] = (bf16_t)v;                                                   \
      if (WRITE_FM) OFb[(size_t)(fb + r) * NODES + n] = (bf16_t)v;         \
    }                                                                      \
    *(v8bf*)(ONb + (size_t)n * Fdim + fb) = ov;                            \
  }
  CHEB_STORE(acc00, 0, 0)
  CHEB_STORE(acc01, 0, 1)
  CHEB_STORE(acc10, 1, 0)
  CHEB_STORE(acc11, 1, 1)
#undef CHEB_STORE
}

// ------------------------------------------------------------- weight GEMM
// out[co,n] = act( sum_f W[co,f]*CX_nm[n,f] + bias[co] ),  K = 3*HID.
// All operands contiguous (node-major B); 2x2 WMMA tiles per wave, no LDS.
template <int OMODE>   // 0: bf16 fm+nm out, relu;  1: f32 node-major out
__global__ __launch_bounds__(128)
void dense_wmma_kernel(const bf16_t* __restrict__ Wbf, const float* __restrict__ bias,
                       const bf16_t* __restrict__ T0, const bf16_t* __restrict__ T1,
                       const bf16_t* __restrict__ T2,   // node-major [B,N,HID]
                       bf16_t* __restrict__ out_fm,     // [B,CO,N]
                       bf16_t* __restrict__ out_nm,     // [B,N,CO]
                       float*  __restrict__ out_f32,    // [B,N,CO]
                       int CO) {
  const int K3 = 3 * HID;
  int wave = (blockIdx.x * 128 + threadIdx.x) >> 5;
  int lane = threadIdx.x & 31;
  int half = lane >> 4, lm = lane & 15;

  int nP = NODES / 32;                      // 4
  int perBatch = (CO / 32) * nP;
  int b  = wave / perBatch;
  int t  = wave % perBatch;
  int co0 = (t / nP) * 32;
  int n0  = (t % nP) * 32;

  const bf16_t* arow0 = Wbf + (size_t)(co0 + lm) * K3 + 8 * half;
  const bf16_t* arow1 = arow0 + (size_t)16 * K3;
  size_t nmBase = ((size_t)b * NODES + n0) * HID;

  v8f acc00 = {}, acc01 = {}, acc10 = {}, acc11 = {};
#pragma unroll 1
  for (int c = 0; c < 3; ++c) {
    const bf16_t* src = ((c == 0) ? T0 : (c == 1) ? T1 : T2) + nmBase;
    const bf16_t* br0 = src + (size_t)lm * HID + 8 * half;
    const bf16_t* br1 = br0 + (size_t)16 * HID;
    int kb = c * HID;
#pragma unroll 4
    for (int k0 = 0; k0 < HID; k0 += 32) {
      v16bf a0 = frag2(arow0 + kb + k0, arow0 + kb + k0 + 16);
      v16bf a1 = frag2(arow1 + kb + k0, arow1 + kb + k0 + 16);
      v16bf b0 = frag2(br0 + k0, br0 + k0 + 16);
      v16bf b1 = frag2(br1 + k0, br1 + k0 + 16);
      acc00 = wmma_bf16(a0, b0, acc00);
      acc01 = wmma_bf16(a0, b1, acc01);
      acc10 = wmma_bf16(a1, b0, acc10);
      acc11 = wmma_bf16(a1, b1, acc11);
    }
  }

#define DENSE_EPI(AC, COI, NI) {                                           \
    int cb = co0 + (COI)*16 + 8 * half;                                    \
    int n  = n0 + (NI)*16 + lm;                                            \
    size_t nmOff = ((size_t)b * NODES + n) * CO + cb;                      \
    if (OMODE == 0) {                                                      \
      v8bf ov;                                                             \
      _Pragma("unroll")                                                    \
      for (int r = 0; r < 8; ++r) {                                        \
        float v = fmaxf((AC)[r] + bias[cb + r], 0.f);                      \
        ov[r] = (bf16_t)v;                                                 \
        out_fm[((size_t)b * CO + cb + r) * NODES + n] = (bf16_t)v;         \
      }                                                                    \
      *(v8bf*)(out_nm + nmOff) = ov;                                       \
    } else {                                                               \
      v4f lo, hi;                                                          \
      _Pragma("unroll")                                                    \
      for (int r = 0; r < 4; ++r) {                                        \
        lo[r] = (AC)[r]     + bias[cb + r];                                \
        hi[r] = (AC)[r + 4] + bias[cb + r + 4];                            \
      }                                                                    \
      *(v4f*)(out_f32 + nmOff)     = lo;                                   \
      *(v4f*)(out_f32 + nmOff + 4) = hi;                                   \
    }                                                                      \
  }
  DENSE_EPI(acc00, 0, 0)
  DENSE_EPI(acc01, 0, 1)
  DENSE_EPI(acc10, 1, 0)
  DENSE_EPI(acc11, 1, 1)
#undef DENSE_EPI
}

// ---------------------------------------------------------------- epilogue
__global__ void pool_kernel(const float* __restrict__ h4, float* __restrict__ pool) {
  int idx = blockIdx.x * 256 + threadIdx.x;   // over B*C_OUT
  if (idx >= BB * C_OUTc) return;
  int b = idx / C_OUTc, c = idx % C_OUTc;
  const float* p = h4 + (size_t)b * NODES * C_OUTc + c;   // node-major
  float s = 0.f;
  for (int n = 0; n < NODES; ++n) s += p[(size_t)n * C_OUTc];
  pool[idx] = s * (1.f / NODES);
}

__global__ void bn_stats_kernel(const float* __restrict__ pool,
                                float* __restrict__ mu, float* __restrict__ rstd) {
  int c = threadIdx.x;                        // 256 threads, 1 block
  float s = 0.f, s2 = 0.f;
  for (int b = 0; b < BB; ++b) {
    float v = pool[b * C_OUTc + c];
    s += v; s2 += v * v;
  }
  float m = s * (1.f / BB);
  float var = s2 * (1.f / BB) - m * m;
  mu[c] = m;
  rstd[c] = rsqrtf(var + 1e-5f);
}

__global__ void fc_kernel(const float* __restrict__ pool, const float* __restrict__ mu,
                          const float* __restrict__ rstd, const float* __restrict__ gamma,
                          const float* __restrict__ beta, const float* __restrict__ fcW,
                          const float* __restrict__ fcb, float* __restrict__ out) {
  int b = blockIdx.x;
  __shared__ float hn[C_OUTc];
  int c = threadIdx.x;                        // 256 threads
  hn[c] = (pool[b * C_OUTc + c] - mu[c]) * rstd[c] * gamma[c] + beta[c];
  __syncthreads();
  if (c < NCc) {
    float s = fcb[c];
    for (int k = 0; k < C_OUTc; ++k) s += hn[k] * fcW[c * C_OUTc + k];
    out[b * NCc + c] = s;
  }
}

// ---------------------------------------------------------------- launcher
extern "C" void kernel_launch(void* const* d_in, const int* in_sizes, int n_in,
                              void* d_out, int out_size, void* d_ws, size_t ws_size,
                              hipStream_t stream) {
  const float* x     = (const float*)d_in[0];
  const float* A     = (const float*)d_in[1];
  const float* W_in  = (const float*)d_in[2];
  const float* b_in  = (const float*)d_in[3];
  const float* W_h0  = (const float*)d_in[4];
  const float* b_h0  = (const float*)d_in[5];
  const float* W_h1  = (const float*)d_in[6];
  const float* b_h1  = (const float*)d_in[7];
  const float* W_out = (const float*)d_in[8];
  const float* b_out = (const float*)d_in[9];
  const float* gamma = (const float*)d_in[10];
  const float* beta  = (const float*)d_in[11];
  const float* fcW   = (const float*)d_in[12];
  const float* fcb   = (const float*)d_in[13];
  float* out = (float*)d_out;

  char* ws = (char*)d_ws;
  size_t off = 0;
  auto alloc = [&](size_t bytes) -> void* {
    void* p = ws + off;
    off += (bytes + 255) & ~(size_t)255;
    return p;
  };
  const size_t featBytes = (size_t)BB * HID * NODES * sizeof(bf16_t);   // 67 MB
  bf16_t* fmH  = (bf16_t*)alloc(featBytes);   // feature-major h (starts as x_fm)
  bf16_t* fmT1 = (bf16_t*)alloc(featBytes);   // feature-major T1
  bf16_t* nm0  = (bf16_t*)alloc(featBytes);   // node-major h ping (starts as x_nm)
  bf16_t* nm1  = (bf16_t*)alloc(featBytes);   // node-major h pong
  bf16_t* t1nm = (bf16_t*)alloc(featBytes);
  bf16_t* t2nm = (bf16_t*)alloc(featBytes);
  bf16_t* Lh   = (bf16_t*)alloc((size_t)BB * NODES * NODES * sizeof(bf16_t));
  bf16_t* Wb0  = (bf16_t*)alloc((size_t)HID   * 3 * F_IN * sizeof(bf16_t));
  bf16_t* Wb1  = (bf16_t*)alloc((size_t)HID   * 3 * HID  * sizeof(bf16_t));
  bf16_t* Wb2  = (bf16_t*)alloc((size_t)HID   * 3 * HID  * sizeof(bf16_t));
  bf16_t* Wb3  = (bf16_t*)alloc((size_t)C_OUTc* 3 * HID  * sizeof(bf16_t));
  float*  h4   = (float*)alloc((size_t)BB * NODES * C_OUTc * sizeof(float));
  float*  pool = (float*)alloc((size_t)BB * C_OUTc * sizeof(float));
  float*  mu   = (float*)alloc(C_OUTc * sizeof(float));
  float*  rstd = (float*)alloc(C_OUTc * sizeof(float));

  auto conv = [&](const float* s, bf16_t* d, int n) {
    f32_to_bf16_kernel<<<(n + 255) / 256, 256, 0, stream>>>(s, d, n);
  };
  conv(W_in,  Wb0, HID * 3 * F_IN);
  conv(W_h0,  Wb1, HID * 3 * HID);
  conv(W_h1,  Wb2, HID * 3 * HID);
  conv(W_out, Wb3, C_OUTc * 3 * HID);

  {
    long long total = (long long)BB * NODES * F_IN;
    x_prep_kernel<<<(unsigned)((total + 255) / 256), 256, 0, stream>>>(x, nm0, fmH);
  }
  normalize_kernel<<<BB, NODES, 0, stream>>>(A, Lh);

  const int chebGrid   = BB * (NODES / 32) * (HID / 128);            // 8192 blocks
  const int denseGridH = BB * (HID    / 32) * (NODES / 32) / 4;      // 8192
  const int denseGridO = BB * (C_OUTc / 32) * (NODES / 32) / 4;      // 4096

  bf16_t* hin_nm = nm0;
  bf16_t* hout_nm = nm1;

  auto layer = [&](const bf16_t* Wb, const float* bv, int CO, bool last) {
    // T1 = L h   (needs fm for next cheb + nm for dense)
    cheb_wmma_kernel<false, true><<<chebGrid, 128, 0, stream>>>(
        fmH, Lh, nullptr, fmT1, t1nm, HID, 1.f, 0.f);
    // T2 = 2 L T1 - h   (nm only)
    cheb_wmma_kernel<true, false><<<chebGrid, 128, 0, stream>>>(
        fmT1, Lh, hin_nm, nullptr, t2nm, HID, 2.f, -1.f);
    if (!last) {
      dense_wmma_kernel<0><<<denseGridH, 128, 0, stream>>>(
          Wb, bv, hin_nm, t1nm, t2nm, fmH, hout_nm, nullptr, CO);
      bf16_t* tmp = hin_nm; hin_nm = hout_nm; hout_nm = tmp;
    } else {
      dense_wmma_kernel<1><<<denseGridO, 128, 0, stream>>>(
          Wb, bv, hin_nm, t1nm, t2nm, nullptr, nullptr, h4, CO);
    }
  };

  layer(Wb0, b_in,  HID,    false);
  layer(Wb1, b_h0,  HID,    false);
  layer(Wb2, b_h1,  HID,    false);
  layer(Wb3, b_out, C_OUTc, true);

  pool_kernel<<<(BB * C_OUTc + 255) / 256, 256, 0, stream>>>(h4, pool);
  bn_stats_kernel<<<1, C_OUTc, 0, stream>>>(pool, mu, rstd);
  fc_kernel<<<BB, C_OUTc, 0, stream>>>(pool, mu, rstd, gamma, beta, fcW, fcb, out);
}
